// MultiHeadSelfAttention_75350906241368
// MI455X (gfx1250) — compile-verified, matrix operand order checked by reading, not run on hardware
//
#include <hip/hip_runtime.h>
#include <hip/hip_bf16.h>

// ---------------------------------------------------------------------------
// MHSA for MI455X (gfx1250, wave32).  All GEMMs + attention on the WMMA units
// in bf16 with fp32 accumulation (v_wmma_f32_16x16x32_bf16).  LDS staging via
// the Tensor Data Mover (tensor_load_to_lds / TENSORcnt) when available,
// double-buffered; pure-b128 manual staging otherwise.  WMMAs grouped for
// A/B operand-reuse hints (OPSEL RA/RB).
// Pipeline:
//   cvt(x) -> bf16 ; cvt_transpose(w*) -> bf16 [N][K]
//   gemm_proj : Q,K = x@w+b  (bf16 [H][S][64]) ; V -> bf16 transposed [H][64][S]
//   attn_flash: per-head online-softmax attention   (bf16 [S][D])
//   gemm_proj : out = attn@wo + bo                  (fp32 -> d_out)
// ---------------------------------------------------------------------------

typedef __bf16 bf16;
typedef __attribute__((ext_vector_type(16))) __bf16 bf16x16;
typedef __attribute__((ext_vector_type(8)))  __bf16 bf16x8;
typedef __attribute__((ext_vector_type(4)))  __bf16 bf16x4;
typedef __attribute__((ext_vector_type(8)))  float  v8f;
typedef __attribute__((ext_vector_type(4)))  unsigned int v4u;
typedef __attribute__((ext_vector_type(8)))  int    v8i;
typedef __attribute__((ext_vector_type(4)))  int    v4i;

#define S_LEN 4096
#define D_MODEL 1024
#define N_HEADS 16
#define HEAD_D 64

// ---------------------------- TDM support ----------------------------------
#if __has_builtin(__builtin_amdgcn_tensor_load_to_lds)
#define USE_TDM 1
#if __has_include(<hip/amd_detail/amd_gfx1250_TDM.h>)
#define TDM_LOAD(g0, g1, g2, g3) \
  __builtin_amdgcn_tensor_load_to_lds(g0, g1, g2, g3, (v8i){0,0,0,0,0,0,0,0}, 0)
#else
#define TDM_LOAD(g0, g1, g2, g3) \
  __builtin_amdgcn_tensor_load_to_lds(g0, g1, g2, g3, 0)
#endif
#else
#define USE_TDM 0
#endif

#if USE_TDM
__device__ inline void tdm_wait() {
#if __has_builtin(__builtin_amdgcn_s_wait_tensorcnt)
  __builtin_amdgcn_s_wait_tensorcnt(0);
#else
  asm volatile("s_wait_tensorcnt 0x0" ::: "memory");
#endif
}

// 2-D bf16 tile DMA: global (row-major, row stride `stride` elems) -> LDS
// (rows packed contiguously).  D# per ISA 8.3/8.4: count=1, type=2,
// data_size=1 (2 bytes), tile_dim0 = row length, tile_dim1 = #rows.
__device__ inline void tdm_load_tile_2d(unsigned lds_off, const void* gptr,
                                        unsigned row_len, unsigned rows,
                                        unsigned long long stride) {
  unsigned long long ga = (unsigned long long)(uintptr_t)gptr;
  v4u g0;
  g0[0] = 1u;                                            // count=1 valid D#
  g0[1] = lds_off;                                       // lds_addr (bytes)
  g0[2] = (unsigned)(ga & 0xFFFFFFFFu);                  // global_addr[31:0]
  g0[3] = (unsigned)((ga >> 32) & 0x01FFFFFFu) | (2u << 30);  // [56:32]+type=2
  v8i g1;
  g1[0] = (int)(1u << 16);                               // data_size=1 (2B)
  g1[1] = (int)((row_len & 0xFFFFu) << 16);              // tensor_dim0[15:0]
  g1[2] = (int)((row_len >> 16) | ((rows & 0xFFFFu) << 16));   // td0hi|td1lo
  g1[3] = (int)((rows >> 16) | ((row_len & 0xFFFFu) << 16));   // td1hi|tile_dim0
  g1[4] = (int)(rows & 0xFFFFu);                         // tile_dim1 (tile_dim2=0)
  g1[5] = (int)(unsigned)(stride & 0xFFFFFFFFu);         // tensor_dim0_stride lo
  g1[6] = (int)(unsigned)((stride >> 32) & 0xFFFFu);     // stride hi
  g1[7] = 0;
  v4i z = {0, 0, 0, 0};
  TDM_LOAD(g0, g1, z, z);
}

__device__ inline unsigned lds_off_of(const void* p) {
  return (unsigned)(uintptr_t)p;   // LDS aperture: flat[31:0] == LDS byte offset
}
#endif

// ---------------------------------------------------------------------------
// Fragment loader (CDNA5 16-bit A/B layout, ISA 7.12.2): lane L holds row
// r = L&15; comps 0..7 -> K = 8*(L>>4)+0..7, comps 8..15 -> +16.
// Source must be k-contiguous per row (stride `ld` elems, 16B aligned).
// ---------------------------------------------------------------------------
__device__ inline bf16x16 load_frag(const bf16* base, int ld) {
  int lane = threadIdx.x & 31;
  int r  = lane & 15;
  int kb = (lane >> 4) << 3;
  const bf16* p = base + r * ld + kb;
  bf16x8 lo = *(const bf16x8*)(p);
  bf16x8 hi = *(const bf16x8*)(p + 16);
  return __builtin_shufflevector(lo, hi, 0,1,2,3,4,5,6,7,8,9,10,11,12,13,14,15);
}

// WMMA with operand reuse hints (RA/RB OPSEL bits).
#define WMMA(a, b, c, ra, rb) \
  __builtin_amdgcn_wmma_f32_16x16x32_bf16(false, (a), false, (b), (short)0, (c), (ra), (rb))

// ---------------------------------------------------------------------------
// fp32 -> bf16 (flat)
// ---------------------------------------------------------------------------
__global__ __launch_bounds__(256) void cvt_f32_bf16(const float* __restrict__ in,
                                                    bf16* __restrict__ out, int n) {
  int idx = (blockIdx.x * 256 + threadIdx.x) * 4;
  if (idx + 3 < n) {
    float4 f = *(const float4*)(in + idx);
    bf16x4 o;
    o[0] = (bf16)f.x; o[1] = (bf16)f.y; o[2] = (bf16)f.z; o[3] = (bf16)f.w;
    *(bf16x4*)(out + idx) = o;
  }
}

// fp32 [K][N] -> bf16 [N][K] (one-time weight transpose, 32x32 LDS tiles)
__global__ __launch_bounds__(256) void cvt_transpose(const float* __restrict__ in,
                                                     bf16* __restrict__ out) {
  __shared__ float tile[32][33];
  int n0 = blockIdx.x * 32, k0 = blockIdx.y * 32;
  int tx = threadIdx.x & 31, ty = threadIdx.x >> 5;
#pragma unroll
  for (int i = 0; i < 32; i += 8)
    tile[ty + i][tx] = in[(size_t)(k0 + ty + i) * D_MODEL + n0 + tx];
  __syncthreads();
#pragma unroll
  for (int i = 0; i < 32; i += 8)
    out[(size_t)(n0 + ty + i) * D_MODEL + k0 + tx] = (bf16)tile[tx][ty + i];
}

// ---------------------------------------------------------------------------
// WMMA GEMM:  C[M,N] = X[M,K] @ WT[N,K]^T + bias[N],  M=4096, N=K=1024.
// Block tile 128x128, BK=32, double-buffered LDS; 8 waves 4(M)x2(N), each
// wave 32x64 via 2x4 WMMA (8 wmma / K-chunk, A-reuse hinted).
// mode 0: bf16 [H][S][64] (Q,K).  mode 1: fp32 [M][N] (final).
// mode 2: bf16 transposed heads [H][64][S] (V) -- packed b128 stores.
// ---------------------------------------------------------------------------
__global__ __launch_bounds__(256) void gemm_proj(const bf16* __restrict__ X,
                                                 const bf16* __restrict__ WT,
                                                 const float* __restrict__ bias,
                                                 bf16* __restrict__ outH,
                                                 float* __restrict__ outF,
                                                 int mode) {
  const int N = D_MODEL, K = D_MODEL;
  __shared__ __attribute__((aligned(16))) bf16 sA[2][128 * 32];  // [m][k]
  __shared__ __attribute__((aligned(16))) bf16 sB[2][128 * 32];  // [n][k]

  int tid = threadIdx.x;
  int wid = tid >> 5;
  int m0 = blockIdx.x * 128;
  int n0 = blockIdx.y * 128;
  int wm = (wid >> 1) * 32;
  int wn = (wid & 1) * 64;

  v8f acc[2][4] = {};

#if USE_TDM
  auto issue = [&](int kc, int b) {
    if (wid == 0) {
      tdm_load_tile_2d(lds_off_of(&sA[b][0]), X  + (size_t)m0 * K + kc * 32,
                       32, 128, K);
      tdm_load_tile_2d(lds_off_of(&sB[b][0]), WT + (size_t)n0 * K + kc * 32,
                       32, 128, K);
    }
  };
  auto finish = [&]() { if (wid == 0) tdm_wait(); };
#else
  auto issue = [&](int kc, int b) {
    int k0 = kc * 32;
    int row = tid >> 1, half = (tid & 1) * 16;
    {   // A tile 128x32
      const bf16* g = X + (size_t)(m0 + row) * K + k0 + half;
      bf16* s = &sA[b][row * 32 + half];
      *(bf16x8*)(s)     = *(const bf16x8*)(g);
      *(bf16x8*)(s + 8) = *(const bf16x8*)(g + 8);
      __builtin_prefetch(g + 32, 0, 1);
    }
    {   // B tile 128x32 from WT
      const bf16* g = WT + (size_t)(n0 + row) * K + k0 + half;
      bf16* s = &sB[b][row * 32 + half];
      *(bf16x8*)(s)     = *(const bf16x8*)(g);
      *(bf16x8*)(s + 8) = *(const bf16x8*)(g + 8);
    }
  };
  auto finish = [&]() {};
#endif

  issue(0, 0);
  finish();
  __syncthreads();

  const int KCH = K / 32;
  for (int kc = 0; kc < KCH; ++kc) {
    int b = kc & 1;
    if (kc + 1 < KCH) issue(kc + 1, b ^ 1);

    bf16x16 a0 = load_frag(&sA[b][(wm +  0) * 32], 32);
    bf16x16 a1 = load_frag(&sA[b][(wm + 16) * 32], 32);
    bf16x16 bf[4];
#pragma unroll
    for (int j = 0; j < 4; ++j)
      bf[j] = load_frag(&sB[b][(wn + j * 16) * 32], 32);

    acc[0][0] = WMMA(a0, bf[0], acc[0][0], false, false);
    acc[0][1] = WMMA(a0, bf[1], acc[0][1], true,  false);   // reuse A
    acc[0][2] = WMMA(a0, bf[2], acc[0][2], true,  false);
    acc[0][3] = WMMA(a0, bf[3], acc[0][3], true,  false);
    acc[1][0] = WMMA(a1, bf[0], acc[1][0], false, false);
    acc[1][1] = WMMA(a1, bf[1], acc[1][1], true,  false);
    acc[1][2] = WMMA(a1, bf[2], acc[1][2], true,  false);
    acc[1][3] = WMMA(a1, bf[3], acc[1][3], true,  false);

    finish();
    __syncthreads();
  }

  // epilogue: C layout -> lane holds column n = lane&15, rows 8*(lane>>4)+c
  int lane  = tid & 31;
  int colL  = lane & 15;
  int rbase = (lane >> 4) * 8;
#pragma unroll
  for (int i = 0; i < 2; ++i) {
#pragma unroll
    for (int j = 0; j < 4; ++j) {
      int n = n0 + wn + j * 16 + colL;
      float bv = bias[n];
      int m = m0 + wm + i * 16 + rbase;
      if (mode == 0) {              // [H][S][64]
        int h = n >> 6, d = n & 63;
#pragma unroll
        for (int c = 0; c < 8; ++c)
          outH[((size_t)h * S_LEN + m + c) * HEAD_D + d] = (bf16)(acc[i][j][c] + bv);
      } else if (mode == 2) {       // V^T: [H][64][S], 8 consecutive m -> b128
        int h = n >> 6, d = n & 63;
        bf16x8 pk;
#pragma unroll
        for (int c = 0; c < 8; ++c) pk[c] = (bf16)(acc[i][j][c] + bv);
        *(bf16x8*)(outH + ((size_t)h * HEAD_D + d) * S_LEN + m) = pk;
      } else {                      // fp32 [M][N]
#pragma unroll
        for (int c = 0; c < 8; ++c)
          outF[(size_t)(m + c) * N + n] = acc[i][j][c] + bv;
      }
    }
  }
}

// ---------------------------------------------------------------------------
// Flash attention: one (head, 64-q-row) tile / block, 4 waves, each wave owns
// 16 q rows.  kv chunks of 64, double-buffered (16 WMMAs / chunk).  Scores
// computed transposed (S^T = K @ Q^T) so the softmax'd C-layout is exactly
// the A-fragment layout of P for the P@V WMMAs (no cross-lane movement).
// ---------------------------------------------------------------------------
__global__ __launch_bounds__(128) void attn_flash(const bf16* __restrict__ Q,
                                                  const bf16* __restrict__ Kh_,
                                                  const bf16* __restrict__ VT_,
                                                  bf16* __restrict__ Out) {
  __shared__ __attribute__((aligned(16))) bf16 sQ[64 * 64];      // [q][d]
  __shared__ __attribute__((aligned(16))) bf16 sK[2][64 * 64];   // [kv][d]
  __shared__ __attribute__((aligned(16))) bf16 sVT[2][64 * 64];  // [d][kv]

  int tid  = threadIdx.x;
  int w    = tid >> 5;
  int lane = tid & 31;
  int h    = blockIdx.x >> 6;
  int q0   = (blockIdx.x & 63) * 64;

  const bf16* Qp  = Q   + ((size_t)h * S_LEN + q0) * HEAD_D;
  const bf16* Kp  = Kh_ +  (size_t)h * S_LEN * HEAD_D;
  const bf16* VTp = VT_ +  (size_t)h * HEAD_D * S_LEN;   // [64][S]

#if USE_TDM
  auto issue = [&](int c, int b) {
    if (w == 0) {
      tdm_load_tile_2d(lds_off_of(&sK[b][0]),  Kp + (size_t)c * 64 * HEAD_D,
                       4096, 1, 4096);                   // contiguous 8KB
      tdm_load_tile_2d(lds_off_of(&sVT[b][0]), VTp + (size_t)c * 64,
                       64, HEAD_D, S_LEN);               // 64 rows x 128B
    }
  };
  auto finish = [&]() { if (w == 0) tdm_wait(); };
#else
  auto issue = [&](int c, int b) {
    int s0 = c * 64;
    int row = tid >> 1, half = (tid & 1) * 32;
    {   // K chunk [kv][d], 64x64
      const bf16* g = Kp + (size_t)(s0 + row) * HEAD_D + half;
      bf16* s = &sK[b][row * 64 + half];
      *(bf16x8*)(s)      = *(const bf16x8*)(g);
      *(bf16x8*)(s + 8)  = *(const bf16x8*)(g + 8);
      *(bf16x8*)(s + 16) = *(const bf16x8*)(g + 16);
      *(bf16x8*)(s + 24) = *(const bf16x8*)(g + 24);
    }
    {   // V^T chunk [d][kv], 64x64
      const bf16* g = VTp + (size_t)row * S_LEN + s0 + half;
      bf16* s = &sVT[b][row * 64 + half];
      *(bf16x8*)(s)      = *(const bf16x8*)(g);
      *(bf16x8*)(s + 8)  = *(const bf16x8*)(g + 8);
      *(bf16x8*)(s + 16) = *(const bf16x8*)(g + 16);
      *(bf16x8*)(s + 24) = *(const bf16x8*)(g + 24);
    }
  };
  auto finish = [&]() {};
#endif

  issue(0, 0);
  {   // stage Q (64x64) once
    int row = tid >> 1, half = (tid & 1) * 32;
    const bf16* g = Qp + row * HEAD_D + half;
    bf16* s = sQ + row * 64 + half;
    *(bf16x8*)(s)      = *(const bf16x8*)(g);
    *(bf16x8*)(s + 8)  = *(const bf16x8*)(g + 8);
    *(bf16x8*)(s + 16) = *(const bf16x8*)(g + 16);
    *(bf16x8*)(s + 24) = *(const bf16x8*)(g + 24);
  }
  finish();
  __syncthreads();

  bf16x16 qlo = load_frag(sQ + (w * 16) * 64,      64);
  bf16x16 qhi = load_frag(sQ + (w * 16) * 64 + 32, 64);

  v8f   o[4] = {};
  float m_run = -__builtin_inff();
  float l_run = 0.f;
  int   srcbase = (lane >> 4) * 8;

  const int NC = S_LEN / 64;
  for (int c = 0; c < NC; ++c) {
    int b = c & 1;
    if (c + 1 < NC) issue(c + 1, b ^ 1);

    // S^T: M = kv (4 tiles of 16), N = q, K-dim = d = 64 (2 steps); B reused
    v8f sv[4] = {};
    {
      bf16x16 ka[4];
#pragma unroll
      for (int t = 0; t < 4; ++t) ka[t] = load_frag(&sK[b][t * 16 * 64], 64);
      sv[0] = WMMA(ka[0], qlo, sv[0], false, false);
      sv[1] = WMMA(ka[1], qlo, sv[1], false, true);    // reuse B
      sv[2] = WMMA(ka[2], qlo, sv[2], false, true);
      sv[3] = WMMA(ka[3], qlo, sv[3], false, true);
    }
    {
      bf16x16 kb[4];
#pragma unroll
      for (int t = 0; t < 4; ++t) kb[t] = load_frag(&sK[b][t * 16 * 64 + 32], 64);
      sv[0] = WMMA(kb[0], qhi, sv[0], false, false);
      sv[1] = WMMA(kb[1], qhi, sv[1], false, true);
      sv[2] = WMMA(kb[2], qhi, sv[2], false, true);
      sv[3] = WMMA(kb[3], qhi, sv[3], false, true);
    }

    // online softmax (lane = one q column; 32 kv values in sv[0..3])
    float cmax = -__builtin_inff();
#pragma unroll
    for (int t = 0; t < 4; ++t)
#pragma unroll
      for (int i = 0; i < 8; ++i) {
        sv[t][i] *= 0.125f;                              // 1/sqrt(64)
        cmax = fmaxf(cmax, sv[t][i]);
      }
    cmax = fmaxf(cmax, __shfl_xor(cmax, 16, 32));
    float m_new = fmaxf(m_run, cmax);
    float corr  = __expf(m_run - m_new);
    float rs = 0.f;
    bf16x16 a_p0, a_p1;   // kv 0..31 / 32..63 as A-fragments (free remap)
#pragma unroll
    for (int i = 0; i < 8; ++i) {
      float p0 = __expf(sv[0][i] - m_new);
      float p1 = __expf(sv[1][i] - m_new);
      float p2 = __expf(sv[2][i] - m_new);
      float p3 = __expf(sv[3][i] - m_new);
      rs += (p0 + p1) + (p2 + p3);
      a_p0[i] = (bf16)p0;  a_p0[i + 8] = (bf16)p1;
      a_p1[i] = (bf16)p2;  a_p1[i + 8] = (bf16)p3;
    }
    rs += __shfl_xor(rs, 16, 32);
    l_run = l_run * corr + rs;
    m_run = m_new;

    float corrv[8];
#pragma unroll
    for (int i = 0; i < 8; ++i) corrv[i] = __shfl(corr, srcbase + i, 32);
#pragma unroll
    for (int i = 0; i < 8; ++i) {
      o[0][i] *= corrv[i]; o[1][i] *= corrv[i];
      o[2][i] *= corrv[i]; o[3][i] *= corrv[i];
    }

    // O += P @ V : K-dim = kv = 64 (2 steps), N = d (4 tiles); A reused
    {
      bf16x16 vf[4];
#pragma unroll
      for (int t = 0; t < 4; ++t) vf[t] = load_frag(&sVT[b][t * 16 * 64], 64);
      o[0] = WMMA(a_p0, vf[0], o[0], false, false);
      o[1] = WMMA(a_p0, vf[1], o[1], true,  false);      // reuse A
      o[2] = WMMA(a_p0, vf[2], o[2], true,  false);
      o[3] = WMMA(a_p0, vf[3], o[3], true,  false);
    }
    {
      bf16x16 vg[4];
#pragma unroll
      for (int t = 0; t < 4; ++t) vg[t] = load_frag(&sVT[b][t * 16 * 64 + 32], 64);
      o[0] = WMMA(a_p1, vg[0], o[0], false, false);
      o[1] = WMMA(a_p1, vg[1], o[1], true,  false);
      o[2] = WMMA(a_p1, vg[2], o[2], true,  false);
      o[3] = WMMA(a_p1, vg[3], o[3], true,  false);
    }

    finish();
    __syncthreads();
  }

  float lv[8];
#pragma unroll
  for (int i = 0; i < 8; ++i) lv[i] = __shfl(l_run, srcbase + i, 32);
#pragma unroll
  for (int nt = 0; nt < 4; ++nt) {
#pragma unroll
    for (int i = 0; i < 8; ++i) {
      int qrow = q0 + w * 16 + srcbase + i;
      int col  = h * HEAD_D + nt * 16 + (lane & 15);
      Out[(size_t)qrow * D_MODEL + col] = (bf16)(o[nt][i] / lv[i]);
    }
  }
}

// ---------------------------------------------------------------------------
// launch
// ---------------------------------------------------------------------------
extern "C" void kernel_launch(void* const* d_in, const int* in_sizes, int n_in,
                              void* d_out, int out_size, void* d_ws, size_t ws_size,
                              hipStream_t stream) {
  (void)in_sizes; (void)n_in; (void)out_size; (void)ws_size;
  const float* x  = (const float*)d_in[0];
  const float* wq = (const float*)d_in[1];
  const float* bq = (const float*)d_in[2];
  const float* wk = (const float*)d_in[3];
  const float* bk = (const float*)d_in[4];
  const float* wv = (const float*)d_in[5];
  const float* bv = (const float*)d_in[6];
  const float* wo = (const float*)d_in[7];
  const float* bo = (const float*)d_in[8];

  const size_t SD = (size_t)S_LEN * D_MODEL;
  const size_t DD = (size_t)D_MODEL * D_MODEL;

  bf16* xb    = (bf16*)d_ws;
  bf16* wqt   = xb  + SD;       // transposed bf16 weights [N][K]
  bf16* wkt   = wqt + DD;
  bf16* wvt   = wkt + DD;
  bf16* wot   = wvt + DD;
  bf16* Qh    = wot + DD;       // [H][S][64]
  bf16* Kh    = Qh  + SD;
  bf16* VTh   = Kh  + SD;       // [H][64][S]
  bf16* attnb = VTh + SD;       // [S][D]   (48 MB total)

  cvt_f32_bf16<<<SD / 4 / 256, 256, 0, stream>>>(x, xb, (int)SD);
  dim3 tgrid(D_MODEL / 32, D_MODEL / 32);
  cvt_transpose<<<tgrid, 256, 0, stream>>>(wq, wqt);
  cvt_transpose<<<tgrid, 256, 0, stream>>>(wk, wkt);
  cvt_transpose<<<tgrid, 256, 0, stream>>>(wv, wvt);
  cvt_transpose<<<tgrid, 256, 0, stream>>>(wo, wot);

  dim3 ggrid(S_LEN / 128, D_MODEL / 128);
  gemm_proj<<<ggrid, 256, 0, stream>>>(xb, wqt, bq, Qh,  nullptr, 0);
  gemm_proj<<<ggrid, 256, 0, stream>>>(xb, wkt, bk, Kh,  nullptr, 0);
  gemm_proj<<<ggrid, 256, 0, stream>>>(xb, wvt, bv, VTh, nullptr, 2);

  attn_flash<<<N_HEADS * (S_LEN / 64), 128, 0, stream>>>(Qh, Kh, VTh, attnb);

  gemm_proj<<<ggrid, 256, 0, stream>>>(attnb, wot, bo, nullptr, (float*)d_out, 1);
}
